// CorrespondenceDecoder_6932077216083
// MI455X (gfx1250) — matrix operand order repaired and checked by reading, not compiled, not run on hardware
//
#include <hip/hip_runtime.h>
#include <hip/hip_bf16.h>
#include <math.h>

typedef __attribute__((ext_vector_type(16))) __bf16 v16bf;
typedef __attribute__((ext_vector_type(8)))  __bf16 v8bf;
typedef __attribute__((ext_vector_type(8)))  float  v8f;

#define S_LEN 2048
#define BATCH 8
#define DIM   512
#define KNBR  64
#define LDS_STRIDE 48   /* 32 bf16 payload + 16 pad; 96B row stride, 32B aligned */

#if defined(__has_builtin)
#if __has_builtin(__builtin_amdgcn_global_load_async_to_lds_b128) && \
    __has_builtin(__builtin_amdgcn_s_wait_asynccnt)
#define USE_ASYNC_LDS 1
#endif
#endif

#ifdef USE_ASYNC_LDS
typedef int async_v4i __attribute__((vector_size(16)));
typedef __attribute__((address_space(1))) async_v4i* async_gptr;  // global (AS1)
typedef __attribute__((address_space(3))) async_v4i* async_lptr;  // LDS    (AS3)
#endif

__device__ __forceinline__ __bf16 f2bf(float f) {
  union { float f; unsigned u; } c; c.f = f;
  unsigned u = c.u;
  unsigned r = u + 0x7FFFu + ((u >> 16) & 1u);   // round-to-nearest-even
  unsigned short h = (unsigned short)(r >> 16);
  union { unsigned short s; __bf16 b; } o; o.s = h;
  return o.b;
}

// Stage 16 bytes (8 bf16) from global into LDS.
__device__ __forceinline__ void stage16B(const __bf16* __restrict__ src,
                                         __bf16* dst) {
#ifdef USE_ASYNC_LDS
  __builtin_amdgcn_global_load_async_to_lds_b128(
      (async_gptr)(unsigned long long)(const void*)src,
      (async_lptr)(unsigned)(unsigned long long)(const void*)dst,
      0, 0);
#else
  *(v8bf*)dst = *(const v8bf*)src;
#endif
}

__device__ __forceinline__ void stage_wait() {
#ifdef USE_ASYNC_LDS
  __builtin_amdgcn_s_wait_asynccnt(0);
#endif
  __syncthreads();
}

// ---------------------------------------------------------------------------
// Elementwise f32 -> bf16 (one-time conversion; keeps GEMM loops WMMA-bound).
// ---------------------------------------------------------------------------
__global__ __launch_bounds__(256)
void f32_to_bf16_kernel(const float* __restrict__ in, __bf16* __restrict__ out,
                        int n4) {
  const int i = (blockIdx.x * 256 + threadIdx.x) * 4;
  if (i < n4 * 4) {
    const float4 v = *(const float4*)(in + i);
    out[i + 0] = f2bf(v.x);
    out[i + 1] = f2bf(v.y);
    out[i + 2] = f2bf(v.z);
    out[i + 3] = f2bf(v.w);
  }
}

// ---------------------------------------------------------------------------
// Projection GEMM (all-bf16 operands):
//   Y[n,j] = (sum_d X[n,d]*W[j,d] + bias[j]) * scale   -> bf16
// Block (256thr/8 waves) computes 128 rows x 64 cols; the 64x32 W tile per
// K-step is staged in LDS (async-to-LDS when available) and shared by all
// 8 waves. Wave w: rows [blk*128 + w*16, +16).
// ---------------------------------------------------------------------------
__global__ __launch_bounds__(256)
void proj_gemm_kernel(const __bf16* __restrict__ X,   // (N,512) bf16
                      const __bf16* __restrict__ W,   // (512,512) bf16
                      const float* __restrict__ bias,
                      float scale,
                      __bf16* __restrict__ Y)
{
  __shared__ alignas(64) __bf16 ldsW[64 * LDS_STRIDE];
  const int tid  = threadIdx.x;
  const int lane = tid & 31;
  const int wave = tid >> 5;
  const int rowbase = blockIdx.x * 128 + wave * 16;
  const int cb = blockIdx.y * 64;
  const int m  = lane & 15;
  const int kh = (lane >> 4) * 8;    // unused; A-frag also loads via kb
  const int kb = (lane >> 4) * 16;   // B-frag K-block select
  const int crow   = tid >> 2;       // staging: 64 rows x 4 chunks
  const int cchunk = (tid & 3) * 8;

  v8f acc[4] = {};
  const __bf16* arow = X + (size_t)(rowbase + m) * DIM;

  for (int kt = 0; kt < DIM; kt += 32) {
    stage16B(W + (size_t)(cb + crow) * DIM + kt + cchunk,
             ldsW + crow * LDS_STRIDE + cchunk);
    stage_wait();

    const __bf16* ap = arow + kt;
    __builtin_prefetch(ap + 32, 0, 1);
    union { v16bf v; v8bf h[2]; } a;
    a.h[0] = *(const v8bf*)(ap + kb);
    a.h[1] = *(const v8bf*)(ap + kb + 8);

    #pragma unroll
    for (int t = 0; t < 4; ++t) {
      const v16bf bb = *(const v16bf*)(ldsW + (t * 16 + m) * LDS_STRIDE + kb);
      acc[t] = __builtin_amdgcn_wmma_f32_16x16x32_bf16(
          false, a.v, false, bb, (short)0, acc[t], false, false);
    }
    __syncthreads();
  }

  #pragma unroll
  for (int t = 0; t < 4; ++t) {
    const int col = cb + t * 16 + (lane & 15);
    const float bv = bias[col];
    #pragma unroll
    for (int r = 0; r < 8; ++r) {
      const int mr = (lane >> 4) * 8 + r;
      Y[(size_t)(rowbase + mr) * DIM + col] = f2bf((acc[t][r] + bv) * scale);
    }
  }
}

// ---------------------------------------------------------------------------
// Attention logits: attn[b,q,s] = sum_d Q[(q*B+b),d] * K[(s*B+b),d]
// Padding baked in: s >= klen[b] -> -inf.
// Block computes 128 q x 64 s for one batch; 64x32 K tile per K-step staged
// in LDS, shared by all 8 waves. Grid (32 s-tiles, 16 q-tiles, 8 batches).
// ---------------------------------------------------------------------------
__global__ __launch_bounds__(256)
void attn_gemm_kernel(const __bf16* __restrict__ Q,
                      const __bf16* __restrict__ Kp,
                      const int* __restrict__ klens,
                      float* __restrict__ attn)
{
  __shared__ alignas(64) __bf16 ldsK[64 * LDS_STRIDE];
  const int tid  = threadIdx.x;
  const int lane = tid & 31;
  const int wave = tid >> 5;
  const int sb = blockIdx.x * 64;
  const int qbase = blockIdx.y * 128 + wave * 16;
  const int b  = blockIdx.z;
  const int kh = (lane >> 4) * 8;
  const int kb = (lane >> 4) * 16;
  const int crow   = tid >> 2;
  const int cchunk = (tid & 3) * 8;
  const int klen = klens[b];

  const __bf16* qrow = Q + ((size_t)(qbase + (lane & 15)) * BATCH + b) * DIM;
  v8f acc[4] = {};

  for (int kt = 0; kt < DIM; kt += 32) {
    stage16B(Kp + ((size_t)(sb + crow) * BATCH + b) * DIM + kt + cchunk,
             ldsK + crow * LDS_STRIDE + cchunk);
    stage_wait();

    const __bf16* ap = qrow + kt;
    __builtin_prefetch(ap + 32, 0, 1);
    union { v16bf v; v8bf h[2]; } a;
    a.h[0] = *(const v8bf*)(ap + kb);
    a.h[1] = *(const v8bf*)(ap + kb + 8);

    #pragma unroll
    for (int t = 0; t < 4; ++t) {
      const v16bf bb = *(const v16bf*)(ldsK + (t * 16 + (lane & 15)) * LDS_STRIDE + kb);
      acc[t] = __builtin_amdgcn_wmma_f32_16x16x32_bf16(
          false, a.v, false, bb, (short)0, acc[t], false, false);
    }
    __syncthreads();
  }

  float* obase = attn + (size_t)b * S_LEN * S_LEN;
  #pragma unroll
  for (int t = 0; t < 4; ++t) {
    const int s = sb + t * 16 + (lane & 15);
    const bool pad = (s >= klen);
    #pragma unroll
    for (int r = 0; r < 8; ++r) {
      const int qq = qbase + (lane >> 4) * 8 + r;
      obase[(size_t)qq * S_LEN + s] = pad ? -__builtin_inff() : acc[t][r];
    }
  }
}

__global__ void zero_mask_kernel(unsigned* __restrict__ m) {
  m[threadIdx.x] = 0u;   // 64 threads
}

// ---------------------------------------------------------------------------
// Per-row exact 64th-largest via bitwise threshold select on monotone
// float->uint keys; union of selected columns -> global bitmask (atomicOr).
// ---------------------------------------------------------------------------
__global__ __launch_bounds__(256)
void topk_kernel(const float* __restrict__ attn, unsigned* __restrict__ umask)
{
  __shared__ unsigned srow[S_LEN];
  __shared__ unsigned lmask[S_LEN / 32];
  __shared__ int scnt;
  const int tid = threadIdx.x;
  const int q = blockIdx.x, b = blockIdx.y;
  const float* row = attn + ((size_t)b * S_LEN + q) * S_LEN;

  for (int i = tid; i < S_LEN; i += 256) {
    union { float f; unsigned u; } c; c.f = row[i];
    unsigned u = c.u;
    srow[i] = (u & 0x80000000u) ? ~u : (u | 0x80000000u);
  }
  if (tid < S_LEN / 32) lmask[tid] = 0u;
  __syncthreads();

  unsigned cur = 0;
  for (int bit = 31; bit >= 0; --bit) {
    const unsigned cand = cur | (1u << bit);
    if (tid == 0) scnt = 0;
    __syncthreads();
    int c = 0;
    for (int i = tid; i < S_LEN; i += 256) c += (srow[i] >= cand) ? 1 : 0;
    if (c) atomicAdd(&scnt, c);
    __syncthreads();
    if (scnt >= KNBR) cur = cand;     // max T with count(u>=T) >= K
    __syncthreads();
  }

  for (int i = tid; i < S_LEN; i += 256)
    if (srow[i] >= cur) atomicOr(&lmask[i >> 5], 1u << (i & 31));
  __syncthreads();
  if (tid < S_LEN / 32 && lmask[tid]) atomicOr(&umask[tid], lmask[tid]);
}

// ---------------------------------------------------------------------------
// Masked softmax over s, then out[q,b,:] = sum_s p[s] * xyz[s,b,:].
// ---------------------------------------------------------------------------
__global__ __launch_bounds__(256)
void softmax_av_kernel(const float* __restrict__ attn,
                       const unsigned* __restrict__ umask,
                       const float* __restrict__ xyz,
                       float* __restrict__ out)
{
  __shared__ float red[256];
  __shared__ unsigned m[S_LEN / 32];
  const int tid = threadIdx.x;
  const int q = blockIdx.x, b = blockIdx.y;
  if (tid < S_LEN / 32) m[tid] = umask[tid];
  __syncthreads();

  const float* row = attn + ((size_t)b * S_LEN + q) * S_LEN;
  float vals[S_LEN / 256];
  float lmax = -__builtin_inff();
  #pragma unroll
  for (int k = 0; k < S_LEN / 256; ++k) {
    const int i = tid + k * 256;
    float v = row[i];
    if (!((m[i >> 5] >> (i & 31)) & 1u)) v = -__builtin_inff();
    vals[k] = v;
    lmax = fmaxf(lmax, v);
  }
  red[tid] = lmax; __syncthreads();
  for (int off = 128; off > 0; off >>= 1) {
    if (tid < off) red[tid] = fmaxf(red[tid], red[tid + off]);
    __syncthreads();
  }
  const float gmax = red[0]; __syncthreads();

  float s0 = 0.f, sx = 0.f, sy = 0.f, sz = 0.f;
  #pragma unroll
  for (int k = 0; k < S_LEN / 256; ++k) {
    const int i = tid + k * 256;
    const float e = __expf(vals[k] - gmax);   // exp(-inf)=0 for masked
    const float* vp = xyz + ((size_t)i * BATCH + b) * 3;
    s0 += e; sx += e * vp[0]; sy += e * vp[1]; sz += e * vp[2];
  }
  float sums[4] = {s0, sx, sy, sz};
  float tot[4];
  for (int j = 0; j < 4; ++j) {
    red[tid] = sums[j]; __syncthreads();
    for (int off = 128; off > 0; off >>= 1) {
      if (tid < off) red[tid] += red[tid + off];
      __syncthreads();
    }
    tot[j] = red[0]; __syncthreads();
  }
  if (tid == 0) {
    float* o = out + ((size_t)q * BATCH + b) * 3;
    const float inv = 1.0f / tot[0];
    o[0] = tot[1] * inv; o[1] = tot[2] * inv; o[2] = tot[3] * inv;
  }
}

// ---------------------------------------------------------------------------
// Overlap head: out[n] = dot(X[n,:], Wc) + bc. One wave per output row.
// ---------------------------------------------------------------------------
__global__ __launch_bounds__(256)
void overlap_kernel(const float* __restrict__ X, const float* __restrict__ Wc,
                    const float* __restrict__ bc, float* __restrict__ out, int n)
{
  const int lane = threadIdx.x & 31;
  const int wid = blockIdx.x * 8 + (threadIdx.x >> 5);
  if (wid >= n) return;
  const float* xr = X + (size_t)wid * DIM;
  float s = 0.f;
  for (int d = lane; d < DIM; d += 32) s += xr[d] * Wc[d];
  for (int off = 16; off > 0; off >>= 1) s += __shfl_down(s, off, 32);
  if (lane == 0) out[wid] = s + bc[0];
}

extern "C" void kernel_launch(void* const* d_in, const int* in_sizes, int n_in,
                              void* d_out, int out_size, void* d_ws, size_t ws_size,
                              hipStream_t stream) {
  const float* src_feats = (const float*)d_in[0];
  const float* tgt_feats = (const float*)d_in[1];
  const float* src_xyz   = (const float*)d_in[2];
  const float* tgt_xyz   = (const float*)d_in[3];
  const int*   src_lens  = (const int*)d_in[4];
  const int*   tgt_lens  = (const int*)d_in[5];
  const float* Wq = (const float*)d_in[6];
  const float* bq = (const float*)d_in[7];
  const float* Wk = (const float*)d_in[8];
  const float* bk = (const float*)d_in[9];
  const float* Wc = (const float*)d_in[10];
  const float* bc = (const float*)d_in[11];
  float* out = (float*)d_out;

  char* ws = (char*)d_ws;
  const size_t N = (size_t)S_LEN * BATCH;                 // 16384 rows
  const size_t featBytes = N * DIM * sizeof(__bf16);      // 16 MiB
  const size_t wBytes = (size_t)DIM * DIM * sizeof(__bf16); // 512 KiB
  size_t off = 0;
  __bf16* Xs_bf = (__bf16*)(ws + off); off += featBytes;
  __bf16* Xt_bf = (__bf16*)(ws + off); off += featBytes;
  __bf16* Wq_bf = (__bf16*)(ws + off); off += wBytes;
  __bf16* Wk_bf = (__bf16*)(ws + off); off += wBytes;
  __bf16* Qs = (__bf16*)(ws + off); off += featBytes;
  __bf16* Ks = (__bf16*)(ws + off); off += featBytes;
  __bf16* Qt = (__bf16*)(ws + off); off += featBytes;
  __bf16* Kt = (__bf16*)(ws + off); off += featBytes;
  float* attn = (float*)(ws + off);                       // 128 MiB, reused
  off += (size_t)BATCH * S_LEN * S_LEN * sizeof(float);
  unsigned* umask = (unsigned*)(ws + off);

  const float scale = 1.0f / sqrtf((float)DIM);
  dim3 blk(256);

  // One-time f32 -> bf16 conversions.
  const int nFeat = (int)(N * DIM);        // 8388608
  const int nW = DIM * DIM;                // 262144
  f32_to_bf16_kernel<<<nFeat / 1024, blk, 0, stream>>>(src_feats, Xs_bf, nFeat / 4);
  f32_to_bf16_kernel<<<nFeat / 1024, blk, 0, stream>>>(tgt_feats, Xt_bf, nFeat / 4);
  f32_to_bf16_kernel<<<nW / 1024, blk, 0, stream>>>(Wq, Wq_bf, nW / 4);
  f32_to_bf16_kernel<<<nW / 1024, blk, 0, stream>>>(Wk, Wk_bf, nW / 4);

  // Projections (q gets the 1/sqrt(D) scale baked in, matching reference).
  dim3 pgrid(128, 8);
  proj_gemm_kernel<<<pgrid, blk, 0, stream>>>(Xs_bf, Wq_bf, bq, scale, Qs);
  proj_gemm_kernel<<<pgrid, blk, 0, stream>>>(Xt_bf, Wq_bf, bq, scale, Qt);
  proj_gemm_kernel<<<pgrid, blk, 0, stream>>>(Xs_bf, Wk_bf, bk, 1.0f, Ks);
  proj_gemm_kernel<<<pgrid, blk, 0, stream>>>(Xt_bf, Wk_bf, bk, 1.0f, Kt);

  // Direction 0: src queries -> tgt keys, value = tgt_xyz -> src_corr.
  dim3 agrid(32, 16, 8);
  attn_gemm_kernel<<<agrid, blk, 0, stream>>>(Qs, Kt, tgt_lens, attn);
  zero_mask_kernel<<<1, S_LEN / 32, 0, stream>>>(umask);
  topk_kernel<<<dim3(S_LEN, BATCH), blk, 0, stream>>>(attn, umask);
  softmax_av_kernel<<<dim3(S_LEN, BATCH), blk, 0, stream>>>(attn, umask, tgt_xyz, out);

  // Direction 1: tgt queries -> src keys, value = src_xyz -> tgt_corr.
  attn_gemm_kernel<<<agrid, blk, 0, stream>>>(Qt, Ks, src_lens, attn);
  zero_mask_kernel<<<1, S_LEN / 32, 0, stream>>>(umask);
  topk_kernel<<<dim3(S_LEN, BATCH), blk, 0, stream>>>(attn, umask);
  softmax_av_kernel<<<dim3(S_LEN, BATCH), blk, 0, stream>>>(attn, umask, src_xyz,
                                                            out + (size_t)S_LEN * BATCH * 3);

  // Overlap heads.
  float* ov = out + 2 * (size_t)S_LEN * BATCH * 3;
  overlap_kernel<<<(unsigned)(N / 8), blk, 0, stream>>>(src_feats, Wc, bc, ov, (int)N);
  overlap_kernel<<<(unsigned)(N / 8), blk, 0, stream>>>(tgt_feats, Wc, bc, ov + N, (int)N);
}
